// TemporalAttention_62088047231243
// MI455X (gfx1250) — matrix operand order repaired and checked by reading, not compiled, not run on hardware
//
#include <hip/hip_runtime.h>

#define HID    128
#define NHEADS 8
#define HEADD  16
#define NFB    64
#define AA     32
#define MM     6
#define TT     50
#define EE     30000
#define NNODE  (AA*MM*TT)   /* 9600  */
#define EMX    (EE*MM)      /* 180000 */
#define TWO_PI_F 6.28318530717958647692f
#define PI_F     3.14159265358979323846f

typedef __bf16 bf16_t;
typedef __attribute__((ext_vector_type(16))) __bf16 v16bf;
typedef __attribute__((ext_vector_type(8)))  float  v8f;

// ---------------------------------------------------------------- WMMA helpers
__device__ __forceinline__ v8f wmma_bf16(v16bf a, v16bf b, v8f c) {
  return __builtin_amdgcn_wmma_f32_16x16x32_bf16(false, a, false, b, (short)0, c,
                                                 false, false);
}
// A fragment (16x32 bf16): lane holds row = lane&15; K groups:
// lanes 0-15 -> K {0..7, 16..23}, lanes 16-31 -> K {8..15, 24..31}
__device__ __forceinline__ v16bf load_a16(const bf16_t* p) {
  v16bf f;
#pragma unroll
  for (int e = 0; e < 8; ++e) { f[e] = p[e]; f[e + 8] = p[e + 16]; }
  return f;
}
// B fragment (32x16 bf16): lane holds col = lane&15; lanes 0-15 K {0..7, 16..23},
// lanes 16-31 K {8..15, 24..31} (from transposed weights)
__device__ __forceinline__ v16bf load_b16(const bf16_t* p) {
  v16bf f;
#pragma unroll
  for (int e = 0; e < 8; ++e) { f[e] = p[e]; f[e + 8] = p[e + 16]; }
  return f;
}
__device__ __forceinline__ const bf16_t* aoff(const bf16_t* t, int stride, int k0, int lane) {
  return t + (size_t)(lane & 15) * stride + k0 + ((lane >> 4) << 3);
}
__device__ __forceinline__ const bf16_t* boff(const bf16_t* WT, int kpad, int n0, int k0, int lane) {
  return WT + (size_t)(n0 + (lane & 15)) * kpad + k0 + ((lane >> 4) << 3);
}
// order-preserving float->uint encoding for atomic max
__device__ __forceinline__ unsigned fenc(float f) {
  unsigned u = __float_as_uint(f);
  return (u & 0x80000000u) ? ~u : (u | 0x80000000u);
}
__device__ __forceinline__ float fdec(unsigned u) {
  unsigned v = (u & 0x80000000u) ? (u & 0x7fffffffu) : ~u;
  return __uint_as_float(v);
}
// wave-parallel LayerNorm stats over a [16][128] f32 LDS tile.
// 8 waves: wave w reduces rows 2w (lanes 0-15) and 2w+1 (lanes 16-31).
// shfl_xor masks 1..8 stay within each 16-lane half of the wave32.
__device__ __forceinline__ void ln_stats(const float (*tile)[128], float* mu, float* rs,
                                         int lane, int wave) {
  int row = 2*wave + (lane >> 4);
  int l = lane & 15;
  float s = 0.f, s2 = 0.f;
#pragma unroll
  for (int e = 0; e < 8; ++e) {
    float v = tile[row][l*8 + e];
    s += v; s2 += v*v;
  }
#pragma unroll
  for (int m = 1; m < 16; m <<= 1) {
    s  += __shfl_xor(s,  m, 32);
    s2 += __shfl_xor(s2, m, 32);
  }
  if (l == 0) {
    float mean = s * (1.f/128.f);
    mu[row] = mean;
    rs[row] = rsqrtf(s2*(1.f/128.f) - mean*mean + 1e-5f);
  }
}

// ------------------------------------------------- weight transpose to bf16 [N][Kpad]
__global__ void wt_transpose_kernel(const float* __restrict__ W, bf16_t* __restrict__ WT,
                                    int K, int Nn, int Kpad) {
  int idx = blockIdx.x * blockDim.x + threadIdx.x;
  int total = Nn * Kpad;
  if (idx >= total) return;
  int n = idx / Kpad, k = idx - n * Kpad;
  float v = (k < K) ? W[(size_t)k * Nn + n] : 0.0f;
  WT[idx] = (bf16_t)v;
}

// ------------------------------------------------- init segment buffers
__global__ void seg_init_kernel(float* agg, unsigned* menc, float* den) {
  int idx = blockIdx.x * blockDim.x + threadIdx.x;
  if (idx < NNODE * HID) agg[idx] = 0.f;
  if (idx < NNODE * NHEADS) { menc[idx] = 0u; den[idx] = 0.f; }
}

// ------------------------------------------------- edge expansion + rel features
__global__ void edges_rel_kernel(const int* __restrict__ edges, const float* __restrict__ pos,
                                 const float* __restrict__ head,
                                 int* __restrict__ e0o, int* __restrict__ e1o,
                                 float* __restrict__ rel) {
  int i = blockIdx.x * blockDim.x + threadIdx.x;
  if (i >= EMX) return;
  int j = i / MM, mi = i - j * MM;
  int r0 = edges[j], r1 = edges[EE + j];
  int n0 = r0 % TT + MM * TT * (r0 / TT) + TT * mi;
  int n1 = r1 % TT + MM * TT * (r1 / TT) + TT * mi;
  e0o[i] = n0; e1o[i] = n1;
  float px = pos[2*n0]   - pos[2*n1];
  float py = pos[2*n0+1] - pos[2*n1+1];
  float dist = sqrtf(px*px + py*py);
  float h0 = head[n0], h1 = head[n1];
  float cx = cosf(h1), cy = sinf(h1);
  float direction = atan2f(cx*py - cy*px, cx*px + cy*py);
  float dh = fmodf(h0 - h1 + PI_F, TWO_PI_F);
  if (dh < 0.f) dh += TWO_PI_F;
  dh -= PI_F;
  float trel = (float)(n0 % TT) - (float)(n1 % TT);
  rel[4*i+0] = dist; rel[4*i+1] = direction; rel[4*i+2] = dh; rel[4*i+3] = trel;
}

// ------------------------------------------------- per-node pre_src/pre_dst LN -> bf16
__global__ void node_ln_kernel(const float* __restrict__ x,
                               const float* bs, const float* gs,
                               const float* bd, const float* gd,
                               bf16_t* __restrict__ xs, bf16_t* __restrict__ xd) {
  int n = blockIdx.x * blockDim.x + threadIdx.x;
  if (n >= NNODE) return;
  const float* xr = x + (size_t)n * HID;
  float s = 0.f, s2 = 0.f;
  for (int c = 0; c < HID; ++c) { float v = xr[c]; s += v; s2 += v*v; }
  float mu = s * (1.f/HID);
  float rs = rsqrtf(s2 * (1.f/HID) - mu*mu + 1e-5f);
  for (int c = 0; c < HID; ++c) {
    float v = (xr[c] - mu) * rs;
    xs[(size_t)n*HID + c] = (bf16_t)(v * gs[c] + bs[c]);
    xd[(size_t)n*HID + c] = (bf16_t)(v * gd[c] + bd[c]);
  }
}

// ------------------------------------------------- generic rows x 128 GEMM (WMMA)
__global__ __launch_bounds__(256)
void gemm_rows_kernel(const bf16_t* __restrict__ inb, const bf16_t* __restrict__ WT,
                      const float* __restrict__ bias, float* __restrict__ out) {
  int base = blockIdx.x * 16;
  int lane = threadIdx.x & 31, wave = threadIdx.x >> 5;
  // prefetch full 32KB weight (256 threads x 128B) while addresses resolve
  __builtin_prefetch((const char*)WT + threadIdx.x * 128, 0, 3);
  const bf16_t* arow = inb + (size_t)base * HID;
  v8f acc = {};
#pragma unroll
  for (int k0 = 0; k0 < 128; k0 += 32)
    acc = wmma_bf16(load_a16(aoff(arow, HID, k0, lane)),
                    load_b16(boff(WT, HID, wave*16, k0, lane)), acc);
  int col = wave*16 + (lane & 15);
  int rb  = (lane >> 4) << 3;
  float bv = bias ? bias[col] : 0.f;
#pragma unroll
  for (int j = 0; j < 8; ++j)
    out[(size_t)(base + j + rb) * HID + col] = acc[j] + bv;
}

// ------------------------------------------------- fused Fourier embedding
struct FourierArgs {
  const float* rel; const float* freqs;
  const bf16_t* l1T[4]; const float* l1b[4];
  const float* lnb[4];  const float* lng[4];
  const bf16_t* l2T[4]; const float* l2b[4];
  const bf16_t* outT; const float* outb;
  const float* olnb;  const float* olng;
  bf16_t* rout;
};

__global__ __launch_bounds__(256)
void fourier_kernel(FourierArgs A) {
  __shared__ bf16_t featbuf[16][160];
  __shared__ bf16_t hbuf[16][128];
  __shared__ float  outf[16][128];
  __shared__ float  relrow[16][4];
  __shared__ float  mu[16], rs[16];
  int base = blockIdx.x * 16;
  int tid = threadIdx.x, lane = tid & 31, wave = tid >> 5;
  int col = wave*16 + (lane & 15);
  int rb  = (lane >> 4) << 3;
  __builtin_prefetch((const char*)A.l1T[0] + tid * 128, 0, 3);
  if (tid < 64) relrow[tid >> 2][tid & 3] = A.rel[(size_t)(base + (tid >> 2))*4 + (tid & 3)];
  for (int idx = tid; idx < 16*32; idx += 256)
    featbuf[idx >> 5][128 + (idx & 31)] = (bf16_t)0.f;   // K pad 129->160
  __syncthreads();
  v8f acc = {};
  for (int i = 0; i < 4; ++i) {
    // build [cos | sin | xi] feature tile
    for (int idx = tid; idx < 16*129; idx += 256) {
      int r = idx / 129, c = idx - r*129;
      float xi = relrow[r][i];
      float val;
      if (c < 64)       val = cosf(xi * A.freqs[i*NFB + c]        * TWO_PI_F);
      else if (c < 128) val = sinf(xi * A.freqs[i*NFB + (c - 64)] * TWO_PI_F);
      else              val = xi;
      featbuf[r][c] = (bf16_t)val;
    }
    __syncthreads();
    v8f a1 = {};
#pragma unroll
    for (int k0 = 0; k0 < 160; k0 += 32)
      a1 = wmma_bf16(load_a16(aoff(&featbuf[0][0], 160, k0, lane)),
                     load_b16(boff(A.l1T[i], 160, wave*16, k0, lane)), a1);
    float b1 = A.l1b[i][col];
#pragma unroll
    for (int j = 0; j < 8; ++j) outf[j + rb][col] = a1[j] + b1;
    __syncthreads();
    ln_stats(outf, mu, rs, lane, wave);
    __syncthreads();
    for (int idx = tid; idx < 2048; idx += 256) {
      int r = idx >> 7, c = idx & 127;
      float v = (outf[r][c] - mu[r]) * rs[r] * A.lng[i][c] + A.lnb[i][c];
      hbuf[r][c] = (bf16_t)fmaxf(v, 0.f);
    }
    __syncthreads();
#pragma unroll
    for (int k0 = 0; k0 < 128; k0 += 32)
      acc = wmma_bf16(load_a16(aoff(&hbuf[0][0], 128, k0, lane)),
                      load_b16(boff(A.l2T[i], 128, wave*16, k0, lane)), acc);
    float b2 = A.l2b[i][col];
#pragma unroll
    for (int j = 0; j < 8; ++j) acc[j] += b2;
    __syncthreads();
  }
  // out_ln -> relu -> out projection
#pragma unroll
  for (int j = 0; j < 8; ++j) outf[j + rb][col] = acc[j];
  __syncthreads();
  ln_stats(outf, mu, rs, lane, wave);
  __syncthreads();
  for (int idx = tid; idx < 2048; idx += 256) {
    int r = idx >> 7, c = idx & 127;
    float v = (outf[r][c] - mu[r]) * rs[r] * A.olng[c] + A.olnb[c];
    hbuf[r][c] = (bf16_t)fmaxf(v, 0.f);
  }
  __syncthreads();
  v8f d = {};
#pragma unroll
  for (int k0 = 0; k0 < 128; k0 += 32)
    d = wmma_bf16(load_a16(aoff(&hbuf[0][0], 128, k0, lane)),
                  load_b16(boff(A.outT, 128, wave*16, k0, lane)), d);
  float bo = A.outb[col];
#pragma unroll
  for (int j = 0; j < 8; ++j)
    A.rout[(size_t)(base + j + rb)*HID + col] = (bf16_t)(d[j] + bo);
}

// ------------------------------------------------- edge attention scores
__global__ __launch_bounds__(256)
void edge_attn_kernel(const bf16_t* __restrict__ rbf, const float* prb, const float* prg,
                      const bf16_t* __restrict__ krT, const bf16_t* __restrict__ vrT,
                      const float* vrb,
                      const float* __restrict__ kglob, const float* __restrict__ vglob,
                      const float* __restrict__ qglob,
                      const int* __restrict__ e0, const int* __restrict__ e1,
                      float* __restrict__ sim, unsigned* __restrict__ menc,
                      bf16_t* __restrict__ vjout) {
  __shared__ float  rf[16][128];
  __shared__ bf16_t rbuf[16][128];
  __shared__ float  kjf[16][128];
  __shared__ int    se0[16], se1[16];
  __shared__ float  mu[16], rs[16];
  int base = blockIdx.x * 16;
  int tid = threadIdx.x, lane = tid & 31, wave = tid >> 5;
  int col = wave*16 + (lane & 15);
  int rb  = (lane >> 4) << 3;
  __builtin_prefetch((const char*)krT + tid * 128, 0, 3);
  if (tid < 16) { se0[tid] = e0[base + tid]; se1[tid] = e1[base + tid]; }
  for (int idx = tid; idx < 2048; idx += 256) {
    int r = idx >> 7, c = idx & 127;
    rf[r][c] = (float)rbf[(size_t)(base + r)*HID + c];
  }
  __syncthreads();
  ln_stats(rf, mu, rs, lane, wave);
  __syncthreads();
  for (int idx = tid; idx < 2048; idx += 256) {
    int r = idx >> 7, c = idx & 127;
    rbuf[r][c] = (bf16_t)((rf[r][c] - mu[r]) * rs[r] * prg[c] + prb[c]);
  }
  __syncthreads();
  v8f ak = {}, av = {};
#pragma unroll
  for (int k0 = 0; k0 < 128; k0 += 32) {
    v16bf a = load_a16(aoff(&rbuf[0][0], 128, k0, lane));
    ak = wmma_bf16(a, load_b16(boff(krT, HID, wave*16, k0, lane)), ak);
    av = wmma_bf16(a, load_b16(boff(vrT, HID, wave*16, k0, lane)), av);
  }
  float bvv = vrb[col];
#pragma unroll
  for (int j = 0; j < 8; ++j) {
    int row = j + rb;
    int src = se0[row];
    kjf[row][col] = ak[j] + kglob[(size_t)src*HID + col];
    float vj = av[j] + bvv + vglob[(size_t)src*HID + col];
    vjout[(size_t)(base + row)*HID + col] = (bf16_t)vj;
  }
  __syncthreads();
  if (tid < 128) {
    int row = tid >> 3, h = tid & 7;
    const float* qr = qglob + (size_t)se1[row]*HID + h*HEADD;
    float d = 0.f;
    for (int c = 0; c < HEADD; ++c) d += qr[c] * kjf[row][h*HEADD + c];
    d *= 0.25f; // HEAD^-0.5
    sim[(size_t)(base + row)*NHEADS + h] = d;
    atomicMax(menc + (size_t)se1[row]*NHEADS + h, fenc(d));
  }
}

// ------------------------------------------------- segment softmax pieces
__global__ void seg_mfin_kernel(const unsigned* __restrict__ menc, float* __restrict__ mf) {
  int idx = blockIdx.x * blockDim.x + threadIdx.x;
  if (idx >= NNODE * NHEADS) return;
  float m = fdec(menc[idx]);
  if (!(m > -1e37f && m < 1e37f)) m = 0.f;   // non-finite (empty segment) -> 0
  mf[idx] = m;
}
__global__ void edge_exp_kernel(float* __restrict__ simex, const int* __restrict__ e1,
                                const float* __restrict__ mf, float* __restrict__ den) {
  int idx = blockIdx.x * blockDim.x + threadIdx.x;
  if (idx >= EMX * NHEADS) return;
  int e = idx >> 3, h = idx & 7;
  int seg = e1[e] * NHEADS + h;
  float ex = expf(simex[idx] - mf[seg]);
  simex[idx] = ex;
  atomicAdd(den + seg, ex);
}
__global__ void edge_agg_kernel(const float* __restrict__ ex, const int* __restrict__ e1,
                                const float* __restrict__ den, const bf16_t* __restrict__ vj,
                                float* __restrict__ agg) {
  int idx = blockIdx.x * blockDim.x + threadIdx.x;
  if (idx >= EMX * NHEADS) return;
  int e = idx >> 3, h = idx & 7;
  int dst = e1[e];
  float w = ex[idx] / fmaxf(den[dst*NHEADS + h], 1e-12f);
  const bf16_t* vp = vj + (size_t)e*HID + h*HEADD;
  float* ap = agg + (size_t)dst*HID + h*HEADD;
#pragma unroll
  for (int c = 0; c < HEADD; ++c) atomicAdd(ap + c, w * (float)vp[c]);
}

// ------------------------------------------------- fused gating + out proj + FF
__global__ __launch_bounds__(256)
void node_out_kernel(const float* __restrict__ agg, const bf16_t* __restrict__ xd,
                     const float* __restrict__ sglob, const float* __restrict__ xin,
                     const bf16_t* gT, const float* gb,
                     const bf16_t* toutT, const float* toutb,
                     const float* pfb, const float* pfg,
                     const bf16_t* ff1T, const float* ff1b,
                     const bf16_t* ff2T, const float* ff2b,
                     float* __restrict__ out) {
  __shared__ float  aggf[16][128];
  __shared__ bf16_t aggbf[16][128];
  __shared__ float  xresf[16][128];
  __shared__ bf16_t hln[16][128];
  __shared__ bf16_t hb[16][128];
  __shared__ float  mu[16], rs[16];
  int base = blockIdx.x * 16;
  int tid = threadIdx.x, lane = tid & 31, wave = tid >> 5;
  int col = wave*16 + (lane & 15);
  int rb  = (lane >> 4) << 3;
  __builtin_prefetch((const char*)gT + tid * 256, 0, 3);
  for (int idx = tid; idx < 2048; idx += 256) {
    int r = idx >> 7, c = idx & 127;
    float v = agg[(size_t)(base + r)*HID + c];
    aggf[r][c] = v; aggbf[r][c] = (bf16_t)v;
  }
  __syncthreads();
  // gate: sigmoid([agg, xd] @ Wg + b)
  v8f ag = {};
#pragma unroll
  for (int k0 = 0; k0 < 128; k0 += 32)
    ag = wmma_bf16(load_a16(aoff(&aggbf[0][0], 128, k0, lane)),
                   load_b16(boff(gT, 256, wave*16, k0, lane)), ag);
#pragma unroll
  for (int k0 = 128; k0 < 256; k0 += 32)
    ag = wmma_bf16(load_a16(aoff(xd + (size_t)base*HID, HID, k0 - 128, lane)),
                   load_b16(boff(gT, 256, wave*16, k0, lane)), ag);
  float gbias = gb[col];
  __syncthreads();  // all reads of aggbf done before overwrite
#pragma unroll
  for (int j = 0; j < 8; ++j) {
    int row = j + rb;
    float gv = 1.f / (1.f + expf(-(ag[j] + gbias)));
    float a0 = aggf[row][col];
    float sv = sglob[(size_t)(base + row)*HID + col];
    aggbf[row][col] = (bf16_t)(a0 + gv * (sv - a0));
  }
  __syncthreads();
  // out projection + residual
  v8f ao = {};
#pragma unroll
  for (int k0 = 0; k0 < 128; k0 += 32)
    ao = wmma_bf16(load_a16(aoff(&aggbf[0][0], 128, k0, lane)),
                   load_b16(boff(toutT, HID, wave*16, k0, lane)), ao);
  float ob = toutb[col];
#pragma unroll
  for (int j = 0; j < 8; ++j) {
    int row = j + rb;
    xresf[row][col] = xin[(size_t)(base + row)*HID + col] + ao[j] + ob;
  }
  __syncthreads();
  // pre_ff LN
  ln_stats(xresf, mu, rs, lane, wave);
  __syncthreads();
  for (int idx = tid; idx < 2048; idx += 256) {
    int r = idx >> 7, c = idx & 127;
    hln[r][c] = (bf16_t)((xresf[r][c] - mu[r]) * rs[r] * pfg[c] + pfb[c]);
  }
  __syncthreads();
  // FF: 128 -> 512 (relu) -> 128, chunked so only a 16x128 h-slice is live
  v8f a2 = {};
  for (int kb = 0; kb < 4; ++kb) {
    v8f hacc = {};
    int nb = kb*128 + wave*16;
#pragma unroll
    for (int k0 = 0; k0 < 128; k0 += 32)
      hacc = wmma_bf16(load_a16(aoff(&hln[0][0], 128, k0, lane)),
                       load_b16(boff(ff1T, 128, nb, k0, lane)), hacc);
    float b1 = ff1b[nb + (lane & 15)];
#pragma unroll
    for (int j = 0; j < 8; ++j)
      hb[j + rb][col] = (bf16_t)fmaxf(hacc[j] + b1, 0.f);
    __syncthreads();
#pragma unroll
    for (int k0 = 0; k0 < 128; k0 += 32)
      a2 = wmma_bf16(load_a16(aoff(&hb[0][0], 128, k0, lane)),
                     load_b16(boff(ff2T, 512, wave*16, kb*128 + k0, lane)), a2);
    __syncthreads();
  }
  float b2 = ff2b[col];
#pragma unroll
  for (int j = 0; j < 8; ++j) {
    int row = j + rb;
    out[(size_t)(base + row)*HID + col] = xresf[row][col] + a2[j] + b2;
  }
}

// ---------------------------------------------------------------- host side
// d_in order (jax pytree: dict keys sorted, lists in order; 'W'<'b', 'b'<'g'):
enum {
  IN_X = 0, IN_POS, IN_HEAD, IN_EDGES,
  P_FF1_W, P_FF1_B, P_FF2_W, P_FF2_B,
  P_PRED_B, P_PRED_G, P_PREF_B, P_PREF_G,
  P_PRER_B, P_PRER_G, P_PRES_B, P_PRES_G,
  P_TOG_W, P_TOG_B, P_TOK_W, P_TOKR_W,
  P_TOOUT_W, P_TOOUT_B, P_TOQ_W, P_TOQ_B,
  P_TOS_W, P_TOS_B, P_TOV_W, P_TOV_B,
  P_TOVR_W, P_TOVR_B,
  P_FREQS,            // 30
  P_MLP0 = 31         // per-mlp: l1.W,l1.b,l2.W,l2.b,ln.b,ln.g (x4)
};
#define P_OUT_W  (P_MLP0 + 24)
#define P_OUT_B  (P_MLP0 + 25)
#define P_OLN_B  (P_MLP0 + 26)
#define P_OLN_G  (P_MLP0 + 27)

extern "C" void kernel_launch(void* const* d_in, const int* in_sizes, int n_in,
                              void* d_out, int out_size, void* d_ws, size_t ws_size,
                              hipStream_t stream) {
  (void)in_sizes; (void)n_in; (void)out_size; (void)ws_size;
  const float* x     = (const float*)d_in[IN_X];
  const float* pos   = (const float*)d_in[IN_POS];
  const float* headp = (const float*)d_in[IN_HEAD];
  const int*   edges = (const int*)d_in[IN_EDGES];
  float* outp = (float*)d_out;

  char* wsb = (char*)d_ws;
  size_t off = 0;
  auto AL = [&](size_t bytes) -> void* {
    void* p = wsb + off;
    off += (bytes + 255) & ~(size_t)255;
    return p;
  };
  // intermediates
  int*    e0    = (int*)AL((size_t)EMX * 4);
  int*    e1    = (int*)AL((size_t)EMX * 4);
  float*  rel   = (float*)AL((size_t)EMX * 4 * 4);
  bf16_t* rbfb  = (bf16_t*)AL((size_t)EMX * HID * 2);
  bf16_t* xs    = (bf16_t*)AL((size_t)NNODE * HID * 2);
  bf16_t* xd    = (bf16_t*)AL((size_t)NNODE * HID * 2);
  float*  qb    = (float*)AL((size_t)NNODE * HID * 4);
  float*  kb    = (float*)AL((size_t)NNODE * HID * 4);
  float*  vb    = (float*)AL((size_t)NNODE * HID * 4);
  float*  sb    = (float*)AL((size_t)NNODE * HID * 4);
  unsigned* menc = (unsigned*)AL((size_t)NNODE * NHEADS * 4);
  float*  mf    = (float*)AL((size_t)NNODE * NHEADS * 4);
  float*  den   = (float*)AL((size_t)NNODE * NHEADS * 4);
  float*  simex = (float*)AL((size_t)EMX * NHEADS * 4);
  bf16_t* vj    = (bf16_t*)AL((size_t)EMX * HID * 2);
  float*  agg   = (float*)AL((size_t)NNODE * HID * 4);
  // transposed bf16 weights
  bf16_t* qT    = (bf16_t*)AL(128*128*2);
  bf16_t* kT    = (bf16_t*)AL(128*128*2);
  bf16_t* vT    = (bf16_t*)AL(128*128*2);
  bf16_t* krT   = (bf16_t*)AL(128*128*2);
  bf16_t* vrT   = (bf16_t*)AL(128*128*2);
  bf16_t* sT    = (bf16_t*)AL(128*128*2);
  bf16_t* toutT = (bf16_t*)AL(128*128*2);
  bf16_t* gT    = (bf16_t*)AL(128*256*2);
  bf16_t* ff1T  = (bf16_t*)AL(512*128*2);
  bf16_t* ff2T  = (bf16_t*)AL(128*512*2);
  bf16_t* l1T[4]; bf16_t* l2T[4];
  for (int i = 0; i < 4; ++i) { l1T[i] = (bf16_t*)AL(128*160*2); l2T[i] = (bf16_t*)AL(128*128*2); }
  bf16_t* outT = (bf16_t*)AL(128*128*2);

  auto TR = [&](int pidx, bf16_t* dst, int K, int Nn, int Kpad) {
    int total = Nn * Kpad;
    wt_transpose_kernel<<<(total + 255)/256, 256, 0, stream>>>(
        (const float*)d_in[pidx], dst, K, Nn, Kpad);
  };
  TR(P_TOQ_W, qT, 128, 128, 128);
  TR(P_TOK_W, kT, 128, 128, 128);
  TR(P_TOV_W, vT, 128, 128, 128);
  TR(P_TOKR_W, krT, 128, 128, 128);
  TR(P_TOVR_W, vrT, 128, 128, 128);
  TR(P_TOS_W, sT, 128, 128, 128);
  TR(P_TOOUT_W, toutT, 128, 128, 128);
  TR(P_TOG_W, gT, 256, 128, 256);
  TR(P_FF1_W, ff1T, 128, 512, 128);
  TR(P_FF2_W, ff2T, 512, 128, 512);
  for (int i = 0; i < 4; ++i) {
    TR(P_MLP0 + 6*i + 0, l1T[i], 129, 128, 160);
    TR(P_MLP0 + 6*i + 2, l2T[i], 128, 128, 128);
  }
  TR(P_OUT_W, outT, 128, 128, 128);

  seg_init_kernel<<<(NNODE*HID)/256, 256, 0, stream>>>(agg, menc, den);
  edges_rel_kernel<<<(EMX + 255)/256, 256, 0, stream>>>(edges, pos, headp, e0, e1, rel);
  node_ln_kernel<<<(NNODE + 255)/256, 256, 0, stream>>>(
      x, (const float*)d_in[P_PRES_B], (const float*)d_in[P_PRES_G],
      (const float*)d_in[P_PRED_B], (const float*)d_in[P_PRED_G], xs, xd);

  gemm_rows_kernel<<<NNODE/16, 256, 0, stream>>>(xd, qT, (const float*)d_in[P_TOQ_B], qb);
  gemm_rows_kernel<<<NNODE/16, 256, 0, stream>>>(xs, kT, nullptr, kb);
  gemm_rows_kernel<<<NNODE/16, 256, 0, stream>>>(xs, vT, (const float*)d_in[P_TOV_B], vb);
  gemm_rows_kernel<<<NNODE/16, 256, 0, stream>>>(xd, sT, (const float*)d_in[P_TOS_B], sb);

  FourierArgs fa;
  fa.rel = rel; fa.freqs = (const float*)d_in[P_FREQS];
  for (int i = 0; i < 4; ++i) {
    fa.l1T[i] = l1T[i];
    fa.l1b[i] = (const float*)d_in[P_MLP0 + 6*i + 1];
    fa.l2T[i] = l2T[i];
    fa.l2b[i] = (const float*)d_in[P_MLP0 + 6*i + 3];
    fa.lnb[i] = (const float*)d_in[P_MLP0 + 6*i + 4];
    fa.lng[i] = (const float*)d_in[P_MLP0 + 6*i + 5];
  }
  fa.outT = outT; fa.outb = (const float*)d_in[P_OUT_B];
  fa.olnb = (const float*)d_in[P_OLN_B]; fa.olng = (const float*)d_in[P_OLN_G];
  fa.rout = rbfb;
  fourier_kernel<<<EMX/16, 256, 0, stream>>>(fa);

  edge_attn_kernel<<<EMX/16, 256, 0, stream>>>(
      rbfb, (const float*)d_in[P_PRER_B], (const float*)d_in[P_PRER_G],
      krT, vrT, (const float*)d_in[P_TOVR_B],
      kb, vb, qb, e0, e1, simex, menc, vj);

  seg_mfin_kernel<<<(NNODE*NHEADS)/256, 256, 0, stream>>>(menc, mf);
  edge_exp_kernel<<<(EMX*NHEADS)/256, 256, 0, stream>>>(simex, e1, mf, den);
  edge_agg_kernel<<<(EMX*NHEADS)/256, 256, 0, stream>>>(simex, e1, den, vj, agg);

  node_out_kernel<<<NNODE/16, 256, 0, stream>>>(
      agg, xd, sb, x,
      gT, (const float*)d_in[P_TOG_B],
      toutT, (const float*)d_in[P_TOOUT_B],
      (const float*)d_in[P_PREF_B], (const float*)d_in[P_PREF_G],
      ff1T, (const float*)d_in[P_FF1_B],
      ff2T, (const float*)d_in[P_FF2_B],
      outp);
}
